// Attn_51505247813740
// MI455X (gfx1250) — matrix operand order repaired and checked by reading, not compiled, not run on hardware
//
#include <hip/hip_runtime.h>
#include <hip/hip_bf16.h>

// ---------------------------------------------------------------------------
// B=32, S=1024, H_IN=1024, H=1024 Bahdanau attention scores + masked softmax.
// bf16 WMMA fused GEMM + tanh + W2-reduction, software-pipelined K loop.
// ---------------------------------------------------------------------------

#define B_  32
#define S_  1024
#define H_  1024

typedef __attribute__((ext_vector_type(16))) __bf16   v16bf;
typedef __attribute__((ext_vector_type(8)))  float    v8f;
typedef __attribute__((ext_vector_type(8)))  unsigned int v8u;

struct A2frag { uint4 x; uint4 y; };   // 32 bytes = 16 bf16

__device__ __forceinline__ unsigned short f2bf(float f) {
    __bf16 h = (__bf16)f;
    return __builtin_bit_cast(unsigned short, h);
}

__device__ __forceinline__ float tanh_hw(float x) {
    float y;
    asm("v_tanh_f32 %0, %1" : "=v"(y) : "v"(x));
    return y;
}

// ---------------------------------------------------------------------------
// Kernel 0a: encoder_outputs (S,B,H) f32 -> enc_bf16 [b][s][k] bf16
// ---------------------------------------------------------------------------
__global__ __launch_bounds__(256) void pack_enc_kernel(
    const float* __restrict__ src, __bf16* __restrict__ dst)
{
    int bx = blockIdx.x;
    int b  = bx >> 10;
    int s  = bx & 1023;
    int t  = threadIdx.x;
    const float4* sp = (const float4*)(src + ((size_t)(s * B_ + b) << 10));
    ushort4*      dp = (ushort4*)(dst + ((size_t)bx << 10));
    float4 f = sp[t];
    ushort4 o;
    o.x = f2bf(f.x); o.y = f2bf(f.y); o.z = f2bf(f.z); o.w = f2bf(f.w);
    dp[t] = o;
}

// ---------------------------------------------------------------------------
// Kernel 0b: W1[:, 0:1024] f32 -> wenc_bf16 [h][k] bf16
// ---------------------------------------------------------------------------
__global__ __launch_bounds__(256) void pack_w1_kernel(
    const float* __restrict__ W1, __bf16* __restrict__ dst)
{
    int h = blockIdx.x;
    int t = threadIdx.x;
    const float4* sp = (const float4*)(W1 + (size_t)h * 2048);
    ushort4*      dp = (ushort4*)(dst + ((size_t)h << 10));
    float4 f = sp[t];
    ushort4 o;
    o.x = f2bf(f.x); o.y = f2bf(f.y); o.z = f2bf(f.z); o.w = f2bf(f.w);
    dp[t] = o;
}

// ---------------------------------------------------------------------------
// Kernel 1: hb2[b,h] = dot(hidden[b,:], W1[h, 1024:2048]) + b1[h]   (f32)
// ---------------------------------------------------------------------------
__global__ __launch_bounds__(256) void hidden_proj_kernel(
    const float* __restrict__ hidden, const float* __restrict__ W1,
    const float* __restrict__ b1, float* __restrict__ hb2)
{
    int gid = blockIdx.x * 256 + threadIdx.x;
    int b = gid >> 10;
    int h = gid & 1023;
    const float4* hv = (const float4*)(hidden + ((size_t)b << 10));
    const float4* wv = (const float4*)(W1 + (size_t)h * 2048 + 1024);
    float s = 0.f;
#pragma unroll 4
    for (int j = 0; j < 256; ++j) {
        float4 a = hv[j], w = wv[j];
        s = fmaf(a.x, w.x, s);
        s = fmaf(a.y, w.y, s);
        s = fmaf(a.z, w.z, s);
        s = fmaf(a.w, w.w, s);
    }
    hb2[gid] = s + b1[h];
}

// ---------------------------------------------------------------------------
// Kernel 2 (main): scores[b,s] = sum_h W2[h]*tanh( (enc.W1encT)[b,s,h] + hb2[b,h] )
//
// grid = B * (S/32) = 1024 blocks, 512 threads = 16 waves.
//   block -> (b, 32 s-rows).  wave w -> h-chunk [w*64, w*64+64) = 4 N-tiles.
//   Each wave: 2 M-frags x 4 N-tiles = 8 f32 accumulators.
//   K loop: 32 steps of 32, fully unrolled, ping-pong prefetch of next step's
//   A/B fragments so global_load_b128s overlap the v_wmma group.
// ---------------------------------------------------------------------------
__global__ __launch_bounds__(512, 1) void attn_main_kernel(
    const __bf16* __restrict__ encb,   // [b][s][k] bf16
    const __bf16* __restrict__ wencb,  // [h][k]    bf16
    const float*  __restrict__ hb2,    // [b][h]
    const float*  __restrict__ W2,     // [h]
    float*        __restrict__ scores) // [b][s]
{
    __shared__ float slds[32];

    const int tid   = threadIdx.x;
    const int wave  = tid >> 5;
    const int lane  = tid & 31;
    const int laneN = lane & 15;   // N column / M row within a 16x16 tile
    const int g     = lane >> 4;   // half-wave group
    const int bx    = blockIdx.x;
    const int b     = bx >> 5;
    const int s0    = (bx & 31) << 5;   // 32 s-rows per block
    const int h0    = wave << 6;        // 64 h per wave

    if (tid < 32) slds[tid] = 0.0f;
    __syncthreads();

    // A fragment sources (contiguous k order, same as B): row (s0 + mi*16 + laneN),
    // lane-group picks K half (+16); two 16B chunks at K and K+8.
    const uint4* abase[2];
#pragma unroll
    for (int mi = 0; mi < 2; ++mi)
        abase[mi] = (const uint4*)(
            encb + ((size_t)b << 20) + ((size_t)(s0 + (mi << 4) + laneN) << 10) + (g << 4));
    // B fragment source: column h = h0 + laneN; lane-group picks K half (+16).
    const v8u* bbase = (const v8u*)(
        wencb + ((size_t)(h0 + laneN) << 10) + (g << 4));

    v8f acc[2][4] = {};          // [mi][n]
    A2frag abuf[2][2];           // [buf][mi]
    v8u    bbuf[2][4];           // [buf][n]

    // prologue: load step 0 into buffer 0
#pragma unroll
    for (int mi = 0; mi < 2; ++mi) {
        abuf[0][mi].x = abase[mi][0];
        abuf[0][mi].y = abase[mi][1];
    }
#pragma unroll
    for (int n = 0; n < 4; ++n)
        bbuf[0][n] = bbase[n * 1024];

#pragma unroll
    for (int st = 0; st < 32; ++st) {
        const int cur = st & 1;
        const int nxt = cur ^ 1;
        if (st < 31) {
            const int k0 = (st + 1) << 5;
#pragma unroll
            for (int mi = 0; mi < 2; ++mi) {
                abuf[nxt][mi].x = abase[mi][(k0 >> 3)];
                abuf[nxt][mi].y = abase[mi][(k0 >> 3) + 1];
            }
#pragma unroll
            for (int n = 0; n < 4; ++n)
                bbuf[nxt][n] = bbase[n * 1024 + (k0 >> 4)];
        }
#pragma unroll
        for (int mi = 0; mi < 2; ++mi) {
            v16bf av = __builtin_bit_cast(v16bf, abuf[cur][mi]);
#pragma unroll
            for (int n = 0; n < 4; ++n) {
                v16bf bv = __builtin_bit_cast(v16bf, bbuf[cur][n]);
                acc[mi][n] = __builtin_amdgcn_wmma_f32_16x16x32_bf16(
                    false, av, false, bv, (short)0, acc[mi][n], false, false);
            }
        }
    }

    // Epilogue: h for this lane in tile n is h0 + n*16 + laneN.
    float hb2v[4], w2v[4];
#pragma unroll
    for (int n = 0; n < 4; ++n) {
        int h = h0 + (n << 4) + laneN;
        hb2v[n] = hb2[(b << 10) + h];
        w2v[n]  = W2[h];
    }

    // C layout: VGPR v -> M = v + 8*g (within tile), N = laneN.
#pragma unroll
    for (int mi = 0; mi < 2; ++mi) {
#pragma unroll
        for (int v = 0; v < 8; ++v) {
            float sum = 0.f;
#pragma unroll
            for (int n = 0; n < 4; ++n)
                sum += w2v[n] * tanh_hw(acc[mi][n][v] + hb2v[n]);
            // reduce across the 16 lanes (different h) of this half-group
            for (int off = 8; off > 0; off >>= 1)
                sum += __shfl_xor(sum, off, 16);
            if (laneN == 0)
                atomicAdd(&slds[(mi << 4) + v + (g << 3)], sum); // ds_add_f32
        }
    }
    __syncthreads();

    if (tid < 32)
        scores[(b << 10) + s0 + tid] = slds[tid];
}

// ---------------------------------------------------------------------------
// Kernel 3: masked softmax over s per b. out[b,0,s]. (b2 is softmax-invariant)
// ---------------------------------------------------------------------------
__global__ __launch_bounds__(256) void softmax_kernel(
    const float* __restrict__ scores, const unsigned char* __restrict__ mask,
    float* __restrict__ out)
{
    const int b = blockIdx.x, t = threadIdx.x;
    const int wid = t >> 5, ln = t & 31;
    __shared__ float red[8];
    __shared__ float bcast;

    float v[4];
    float mx = -INFINITY;
#pragma unroll
    for (int i = 0; i < 4; ++i) {
        int s = t + (i << 8);
        float sc = scores[(b << 10) + s];
        v[i] = mask[(b << 10) + s] ? -INFINITY : sc;
        mx = fmaxf(mx, v[i]);
    }
    for (int off = 16; off > 0; off >>= 1)
        mx = fmaxf(mx, __shfl_xor(mx, off, 32));
    if (ln == 0) red[wid] = mx;
    __syncthreads();
    if (t == 0) {
        float m = red[0];
        for (int w = 1; w < 8; ++w) m = fmaxf(m, red[w]);
        bcast = m;
    }
    __syncthreads();
    const float gmax = bcast;

    float e[4];
    float sum = 0.f;
#pragma unroll
    for (int i = 0; i < 4; ++i) {
        e[i] = __expf(v[i] - gmax);   // exp(-inf) = 0 for masked lanes
        sum += e[i];
    }
    for (int off = 16; off > 0; off >>= 1)
        sum += __shfl_xor(sum, off, 32);
    if (ln == 0) red[wid] = sum;
    __syncthreads();
    if (t == 0) {
        float sm = 0.f;
        for (int w = 0; w < 8; ++w) sm += red[w];
        bcast = sm;
    }
    __syncthreads();
    const float inv = 1.0f / bcast;

#pragma unroll
    for (int i = 0; i < 4; ++i)
        out[(b << 10) + t + (i << 8)] = e[i] * inv;
}

// ---------------------------------------------------------------------------
// Launcher.  Inputs: hidden, encoder_outputs, mask, W1, b1, W2, b2
// Workspace: [0,64MiB) enc_bf16 | +2MiB wenc_bf16 | +128KiB hb2 | +128KiB scores
// ---------------------------------------------------------------------------
extern "C" void kernel_launch(void* const* d_in, const int* in_sizes, int n_in,
                              void* d_out, int out_size, void* d_ws, size_t ws_size,
                              hipStream_t stream) {
    const float*         hidden = (const float*)d_in[0];
    const float*         enc    = (const float*)d_in[1];
    const unsigned char* mask   = (const unsigned char*)d_in[2];
    const float*         W1     = (const float*)d_in[3];
    const float*         b1     = (const float*)d_in[4];
    const float*         W2     = (const float*)d_in[5];
    // b2 (d_in[6]) unused: constant shift cancels in softmax.

    char* ws = (char*)d_ws;
    __bf16* enc_bf  = (__bf16*)(ws);
    __bf16* wenc_bf = (__bf16*)(ws + (size_t)67108864);
    float*  hb2     = (float*)(ws + (size_t)69206016);
    float*  scores  = (float*)(ws + (size_t)69337088);
    float*  out     = (float*)d_out;

    pack_enc_kernel   <<<B_ * S_,       256, 0, stream>>>(enc, enc_bf);
    pack_w1_kernel    <<<H_,            256, 0, stream>>>(W1, wenc_bf);
    hidden_proj_kernel<<<B_ * H_ / 256, 256, 0, stream>>>(hidden, W1, b1, hb2);
    attn_main_kernel  <<<B_ * (S_/32),  512, 0, stream>>>(enc_bf, wenc_bf, hb2, W2, scores);
    softmax_kernel    <<<B_,            256, 0, stream>>>(scores, mask, out);
}
